// BootstrapEnsemble_16621523436001
// MI455X (gfx1250) — matrix operand order repaired and checked by reading, not compiled, not run on hardware
//
#include <hip/hip_runtime.h>
#include <hip/hip_bf16.h>
#include <math.h>

typedef __attribute__((ext_vector_type(16))) _Float16 v16h;
typedef __attribute__((ext_vector_type(8)))  _Float16 v8h;
typedef __attribute__((ext_vector_type(4)))  _Float16 v4h;
typedef __attribute__((ext_vector_type(8)))  float    v8f;
typedef __attribute__((ext_vector_type(4)))  unsigned int v4u;
typedef __attribute__((ext_vector_type(8)))  int      v8i;
typedef __attribute__((ext_vector_type(4)))  int      v4i;

#define EM   100      // n_models
#define ED   16       // input dim
#define EH   128      // hidden dim
#define ENH  4        // n hidden layers
#define EN   16384    // batch
#define ROWS 256      // batch rows per block
#define HST  136      // padded LDS row stride (halves): 128 + 4 DWORDs TDM pad
#define XST  40       // padded LDS row stride (halves): 32 + 4 DWORDs TDM pad

// f16 workspace segments (element counts)
#define WHE (EM*ENH*EH*EH)   // 6,553,600  Wh as-is
#define W1E (EM*EH*32)       //   409,600  W1 padded K16->32
#define HDE (EM*16*EH)       //   204,800  head tile: row0=Wmu,row1=Wsig,rest 0
#define XE  (EN*32)          //   524,288  x padded K16->32
#define TOTE (WHE+W1E+HDE+XE)

// group1 word0: data_size=2B (code 1), pad_enable, pad_amount=4 DW (code 3)
#define PADW_W ((1u<<16)|(1u<<20)|(5u<<22)|(3u<<25))  // interval 64 DW (128 halves/row)
#define PADW_X ((1u<<16)|(1u<<20)|(3u<<22)|(3u<<25))  // interval 16 DW (32 halves/row)

// Load a 16-half WMMA fragment from two 16B-aligned LDS locations.
__device__ __forceinline__ v16h ld16(const _Float16* p, const _Float16* q) {
  union { v16h v; v8h h[2]; } u;
  u.h[0] = *(const v8h*)p;
  u.h[1] = *(const v8h*)q;
  return u.v;
}

__device__ __forceinline__ v8f wmma_f16(v16h a, v16h b, v8f c) {
  return __builtin_amdgcn_wmma_f32_16x16x32_f16(false, a, false, b, (short)0, c,
                                                false, false);
}

// Issue one TDM 2D tile copy (global f16 -> LDS with hardware row padding).
// Whole-wave op; caller must ensure exactly one wave executes it.
__device__ __forceinline__ void tdm_load_2d(unsigned lds_addr, const void* gp,
                                            unsigned dim0, unsigned dim1,
                                            unsigned tile0, unsigned tile1,
                                            unsigned w0pad) {
  unsigned long long ga = (unsigned long long)(uintptr_t)gp;
  v4u g0;
  g0[0] = 1u;                                            // count=1 user descriptor
  g0[1] = lds_addr;                                      // LDS byte address
  g0[2] = (unsigned)(ga & 0xFFFFFFFFull);                // global_addr[31:0]
  g0[3] = (unsigned)((ga >> 32) & 0x1FFFFFFull) | (2u << 30); // addr[56:32] | type=2
  v8i g1;
  g1[0] = (int)w0pad;                                    // mask=0,dsize,pad cfg
  g1[1] = (int)((dim0 & 0xFFFFu) << 16);                 // abar=0 | tensor_dim0.lo
  g1[2] = (int)((dim0 >> 16) | ((dim1 & 0xFFFFu) << 16));// dim0.hi | dim1.lo
  g1[3] = (int)((dim1 >> 16) | (tile0 << 16));           // dim1.hi | tile_dim0
  g1[4] = (int)tile1;                                    // tile_dim1 | tile_dim2=0
  g1[5] = (int)dim0;                                     // tensor_dim0_stride.lo
  g1[6] = 0;                                             // stride0.hi | stride1.lo
  g1[7] = 0;
  v4i z4 = {0, 0, 0, 0};
#if defined(__clang_major__) && (__clang_major__ >= 23)
  v8i z8 = {0, 0, 0, 0, 0, 0, 0, 0};
  __builtin_amdgcn_tensor_load_to_lds(g0, g1, z4, z4, z8, 0);
#else
  __builtin_amdgcn_tensor_load_to_lds(g0, g1, z4, z4, 0);
#endif
}

// ---------------- prepass: fp32 -> f16 weight/x images in workspace ------------
__global__ __launch_bounds__(256) void prep_kernel(
    const float* __restrict__ x,   const float* __restrict__ W1,
    const float* __restrict__ Wmu, const float* __restrict__ Wsig,
    const float* __restrict__ Wh,
    _Float16* __restrict__ whH, _Float16* __restrict__ w1H,
    _Float16* __restrict__ headH, _Float16* __restrict__ xH)
{
  int i = blockIdx.x * 256 + threadIdx.x;
  if (i < WHE) { whH[i] = (_Float16)Wh[i]; return; }
  i -= WHE;
  if (i < W1E) {
    int c = i & 31, r = i >> 5;
    w1H[i] = (c < ED) ? (_Float16)W1[r * ED + c] : (_Float16)0.0f;
    return;
  }
  i -= W1E;
  if (i < HDE) {
    int c = i & 127, r = (i >> 7) & 15, mm = i >> 11;
    float v = (r == 0) ? Wmu[mm * EH + c] : (r == 1) ? Wsig[mm * EH + c] : 0.0f;
    headH[i] = (_Float16)v;
    return;
  }
  i -= HDE;
  if (i < XE) {
    int c = i & 31, r = i >> 5;
    xH[i] = (c < ED) ? (_Float16)x[r * ED + c] : (_Float16)0.0f;
  }
}

// ---------------- main kernel: TDM-fed, double-buffered weights ----------------
__global__ __launch_bounds__(256) void ensemble_mlp_tdm_kernel(
    const float* __restrict__ b1,  const float* __restrict__ bh,
    const float* __restrict__ bmu, const float* __restrict__ bsig,
    const _Float16* __restrict__ whH,  const _Float16* __restrict__ w1H,
    const _Float16* __restrict__ headH,const _Float16* __restrict__ xH,
    float* __restrict__ out)
{
  __shared__ __align__(16) _Float16 hS[ROWS * HST];      // activations, 68 KB
  __shared__ __align__(16) _Float16 wBuf[2][EH * HST];   // weight double buffer

  const int tid  = threadIdx.x;
  const int wave = tid >> 5;
  const int lane = tid & 31;
  const int ln16 = lane & 15;
  const int rsel = lane >> 4;
  const int m    = blockIdx.y;
  const int rowBase = blockIdx.x * ROWS;
  const int r0   = wave * 32;

  _Float16* xS  = wBuf[1];                 // layer-1 operands live in wBuf[1]
  _Float16* w1S = wBuf[1] + ROWS * XST;    // 10240 + 5120 halves <= 17408

  if (wave == 0) {
    tdm_load_2d((unsigned)(uintptr_t)xS,  xH + (size_t)rowBase * 32,
                32, ROWS, 32, ROWS, PADW_X);
    tdm_load_2d((unsigned)(uintptr_t)w1S, w1H + (size_t)m * EH * 32,
                32, EH, 32, EH, PADW_X);
    tdm_load_2d((unsigned)(uintptr_t)&wBuf[0][0], whH + (size_t)m * ENH * EH * EH,
                EH, EH, EH, EH, PADW_W);
    __builtin_amdgcn_s_wait_tensorcnt(1);  // x + W1 landed; Wh[0] may still fly
  }
  __syncthreads();

  v8f acc[2][8];

  // ---- layer 1: h = relu(x @ W1^T + b1), K padded 16->32 ----
  {
    int r = r0 + ln16;
    v16h a0 = ld16(&xS[r * XST + rsel * 8], &xS[r * XST + rsel * 8 + 16]);
    v16h a1 = ld16(&xS[(r + 16) * XST + rsel * 8], &xS[(r + 16) * XST + rsel * 8 + 16]);
#pragma unroll
    for (int nt = 0; nt < 8; ++nt) {
      int cr = nt * 16 + ln16;
      v16h b = ld16(&w1S[cr * XST + rsel * 8], &w1S[cr * XST + rsel * 8 + 16]);
      v8f z = {};
      acc[0][nt] = wmma_f16(a0, b, z);
      acc[1][nt] = wmma_f16(a1, b, z);
    }
    const float* bp = b1 + m * EH;
#pragma unroll
    for (int nt = 0; nt < 8; ++nt) {
      float bv = bp[nt * 16 + ln16];
#pragma unroll
      for (int s = 0; s < 2; ++s)
#pragma unroll
        for (int v = 0; v < 8; ++v) {
          float f = fmaxf(acc[s][nt][v] + bv, 0.0f);
          hS[(r0 + s * 16 + v + rsel * 8) * HST + nt * 16 + ln16] = (_Float16)f;
        }
    }
  }
  if (wave == 0) __builtin_amdgcn_s_wait_tensorcnt(0);   // Wh[0] done
  __syncthreads();

  int cur = 0;
  // ---- 4 hidden layers, TDM for layer li+1 overlapped with compute of li ----
  for (int li = 0; li < ENH; ++li) {
    if (wave == 0) {
      if (li + 1 < ENH)
        tdm_load_2d((unsigned)(uintptr_t)&wBuf[cur ^ 1][0],
                    whH + (size_t)(m * ENH + li + 1) * EH * EH,
                    EH, EH, EH, EH, PADW_W);
      else
        tdm_load_2d((unsigned)(uintptr_t)&wBuf[cur ^ 1][0],
                    headH + (size_t)m * 16 * EH,
                    EH, 16, EH, 16, PADW_W);
    }
    const _Float16* wS = wBuf[cur];

    v16h A0[4], A1[4];
#pragma unroll
    for (int kb = 0; kb < 4; ++kb) {
      int r = r0 + ln16;
      A0[kb] = ld16(&hS[r * HST + kb * 32 + rsel * 8],
                    &hS[r * HST + kb * 32 + rsel * 8 + 16]);
      r += 16;
      A1[kb] = ld16(&hS[r * HST + kb * 32 + rsel * 8],
                    &hS[r * HST + kb * 32 + rsel * 8 + 16]);
    }
#pragma unroll
    for (int nt = 0; nt < 8; ++nt) {
      v8f z = {};
      acc[0][nt] = z; acc[1][nt] = z;
      int cr = nt * 16 + ln16;
#pragma unroll
      for (int kb = 0; kb < 4; ++kb) {
        v16h b = ld16(&wS[cr * HST + kb * 32 + rsel * 8],
                      &wS[cr * HST + kb * 32 + rsel * 8 + 16]);
        acc[0][nt] = wmma_f16(A0[kb], b, acc[0][nt]);
        acc[1][nt] = wmma_f16(A1[kb], b, acc[1][nt]);
      }
    }
    const float* bp = bh + (size_t)(m * ENH + li) * EH;
#pragma unroll
    for (int nt = 0; nt < 8; ++nt) {
      float bv = bp[nt * 16 + ln16];
#pragma unroll
      for (int s = 0; s < 2; ++s)
#pragma unroll
        for (int v = 0; v < 8; ++v) {
          float f = fmaxf(acc[s][nt][v] + bv, 0.0f);
          hS[(r0 + s * 16 + v + rsel * 8) * HST + nt * 16 + ln16] = (_Float16)f;
        }
    }
    if (wave == 0) __builtin_amdgcn_s_wait_tensorcnt(0);
    __syncthreads();
    cur ^= 1;
  }

  // ---- heads: col 0 = mu, col 1 = sigma (B rows 2..15 are zero) ----
  {
    const _Float16* wS = wBuf[cur];
    v16h A0[4], A1[4];
#pragma unroll
    for (int kb = 0; kb < 4; ++kb) {
      int r = r0 + ln16;
      A0[kb] = ld16(&hS[r * HST + kb * 32 + rsel * 8],
                    &hS[r * HST + kb * 32 + rsel * 8 + 16]);
      r += 16;
      A1[kb] = ld16(&hS[r * HST + kb * 32 + rsel * 8],
                    &hS[r * HST + kb * 32 + rsel * 8 + 16]);
    }
    v8f c0 = {}, c1 = {};
#pragma unroll
    for (int kb = 0; kb < 4; ++kb) {
      v16h b = ld16(&wS[ln16 * HST + kb * 32 + rsel * 8],
                    &wS[ln16 * HST + kb * 32 + rsel * 8 + 16]);
      c0 = wmma_f16(A0[kb], b, c0);
      c1 = wmma_f16(A1[kb], b, c1);
    }
    if (ln16 < 2) {
      bool isMu = (ln16 == 0);
      float bb  = isMu ? bmu[m] : bsig[m];
      size_t base = (isMu ? (size_t)0 : (size_t)EM * EN) + (size_t)m * EN;
#pragma unroll
      for (int s = 0; s < 2; ++s)
#pragma unroll
        for (int v = 0; v < 8; ++v) {
          float val = (s ? c1[v] : c0[v]) + bb;
          int row = rowBase + r0 + s * 16 + v + rsel * 8;
          out[base + row] = isMu ? val : __expf(val);
        }
    }
  }
}

// ---------------- fallback (self-contained, no workspace): round-1 kernel ------
__global__ __launch_bounds__(256) void ensemble_mlp_kernel(
    const float* __restrict__ x,   const float* __restrict__ W1,  const float* __restrict__ b1,
    const float* __restrict__ Wh,  const float* __restrict__ bh,
    const float* __restrict__ Wmu, const float* __restrict__ bmu,
    const float* __restrict__ Wsig,const float* __restrict__ bsig,
    float* __restrict__ out)
{
  __shared__ __align__(16) _Float16 hS[ROWS * HST];
  __shared__ __align__(16) _Float16 wS[EH * HST];
  __shared__ float biasS[EH];

  const int tid  = threadIdx.x;
  const int wave = tid >> 5;
  const int lane = tid & 31;
  const int ln16 = lane & 15;
  const int rsel = lane >> 4;
  const int m    = blockIdx.y;
  const int rowBase = blockIdx.x * ROWS;
  const int r0   = wave * 32;

  _Float16* xS  = wS;
  _Float16* w1S = wS + ROWS * XST;

  {
    const float4* xs4 = (const float4*)(x + (size_t)rowBase * ED);
#pragma unroll
    for (int j = 0; j < 4; ++j) {
      float4 f = xs4[j * 256 + tid];
      int e = (j * 256 + tid) * 4;
      int r = e >> 4, c = e & 15;
      v4h t = {(_Float16)f.x, (_Float16)f.y, (_Float16)f.z, (_Float16)f.w};
      *(v4h*)&xS[r * XST + c] = t;
    }
    const float4* w4 = (const float4*)(W1 + (size_t)m * EH * ED);
#pragma unroll
    for (int j = 0; j < 2; ++j) {
      float4 f = w4[j * 256 + tid];
      int e = (j * 256 + tid) * 4;
      int r = e >> 4, c = e & 15;
      v4h t = {(_Float16)f.x, (_Float16)f.y, (_Float16)f.z, (_Float16)f.w};
      *(v4h*)&w1S[r * XST + c] = t;
    }
    v8h z8 = {};
    *(v8h*)&xS[tid * XST + 16] = z8;
    *(v8h*)&xS[tid * XST + 24] = z8;
    if (tid < EH) {
      *(v8h*)&w1S[tid * XST + 16] = z8;
      *(v8h*)&w1S[tid * XST + 24] = z8;
      biasS[tid] = b1[m * EH + tid];
    }
  }
  __syncthreads();

  v8f acc[2][8];
  {
    int r = r0 + ln16;
    v16h a0 = ld16(&xS[r * XST + rsel * 8], &xS[r * XST + rsel * 8 + 16]);
    v16h a1 = ld16(&xS[(r + 16) * XST + rsel * 8], &xS[(r + 16) * XST + rsel * 8 + 16]);
#pragma unroll
    for (int nt = 0; nt < 8; ++nt) {
      int cr = nt * 16 + ln16;
      v16h b = ld16(&w1S[cr * XST + rsel * 8], &w1S[cr * XST + rsel * 8 + 16]);
      v8f z = {};
      acc[0][nt] = wmma_f16(a0, b, z);
      acc[1][nt] = wmma_f16(a1, b, z);
    }
#pragma unroll
    for (int nt = 0; nt < 8; ++nt) {
      float bv = biasS[nt * 16 + ln16];
#pragma unroll
      for (int s = 0; s < 2; ++s)
#pragma unroll
        for (int v = 0; v < 8; ++v) {
          float f = fmaxf(acc[s][nt][v] + bv, 0.0f);
          hS[(r0 + s * 16 + v + rsel * 8) * HST + nt * 16 + ln16] = (_Float16)f;
        }
    }
  }

  for (int li = 0; li < ENH; ++li) {
    __syncthreads();
    {
      const float4* w4 = (const float4*)(Wh + (size_t)(m * ENH + li) * EH * EH);
#pragma unroll
      for (int j = 0; j < 16; ++j) {
        float4 f = w4[j * 256 + tid];
        int e = (j * 256 + tid) * 4;
        int r = e >> 7, c = e & 127;
        v4h t = {(_Float16)f.x, (_Float16)f.y, (_Float16)f.z, (_Float16)f.w};
        *(v4h*)&wS[r * HST + c] = t;
      }
      if (tid < EH) biasS[tid] = bh[(size_t)(m * ENH + li) * EH + tid];
    }
    __syncthreads();

    v16h A0[4], A1[4];
#pragma unroll
    for (int kb = 0; kb < 4; ++kb) {
      int r = r0 + ln16;
      A0[kb] = ld16(&hS[r * HST + kb * 32 + rsel * 8],
                    &hS[r * HST + kb * 32 + rsel * 8 + 16]);
      r += 16;
      A1[kb] = ld16(&hS[r * HST + kb * 32 + rsel * 8],
                    &hS[r * HST + kb * 32 + rsel * 8 + 16]);
    }
#pragma unroll
    for (int nt = 0; nt < 8; ++nt) {
      v8f z = {};
      acc[0][nt] = z; acc[1][nt] = z;
      int cr = nt * 16 + ln16;
#pragma unroll
      for (int kb = 0; kb < 4; ++kb) {
        v16h b = ld16(&wS[cr * HST + kb * 32 + rsel * 8],
                      &wS[cr * HST + kb * 32 + rsel * 8 + 16]);
        acc[0][nt] = wmma_f16(A0[kb], b, acc[0][nt]);
        acc[1][nt] = wmma_f16(A1[kb], b, acc[1][nt]);
      }
    }
#pragma unroll
    for (int nt = 0; nt < 8; ++nt) {
      float bv = biasS[nt * 16 + ln16];
#pragma unroll
      for (int s = 0; s < 2; ++s)
#pragma unroll
        for (int v = 0; v < 8; ++v) {
          float f = fmaxf(acc[s][nt][v] + bv, 0.0f);
          hS[(r0 + s * 16 + v + rsel * 8) * HST + nt * 16 + ln16] = (_Float16)f;
        }
    }
  }

  __syncthreads();
  {
    v8h z8 = {};
    if (tid < 224) {
      int r = 2 + (tid >> 4), c = (tid & 15) * 8;
      *(v8h*)&wS[r * HST + c] = z8;
    }
    if (tid < EH) {
      wS[0 * HST + tid] = (_Float16)Wmu[m * EH + tid];
      wS[1 * HST + tid] = (_Float16)Wsig[m * EH + tid];
    }
  }
  __syncthreads();
  {
    v16h A0[4], A1[4];
#pragma unroll
    for (int kb = 0; kb < 4; ++kb) {
      int r = r0 + ln16;
      A0[kb] = ld16(&hS[r * HST + kb * 32 + rsel * 8],
                    &hS[r * HST + kb * 32 + rsel * 8 + 16]);
      r += 16;
      A1[kb] = ld16(&hS[r * HST + kb * 32 + rsel * 8],
                    &hS[r * HST + kb * 32 + rsel * 8 + 16]);
    }
    v8f c0 = {}, c1 = {};
#pragma unroll
    for (int kb = 0; kb < 4; ++kb) {
      v16h b = ld16(&wS[ln16 * HST + kb * 32 + rsel * 8],
                    &wS[ln16 * HST + kb * 32 + rsel * 8 + 16]);
      c0 = wmma_f16(A0[kb], b, c0);
      c1 = wmma_f16(A1[kb], b, c1);
    }
    if (ln16 < 2) {
      bool isMu = (ln16 == 0);
      float bb  = isMu ? bmu[m] : bsig[m];
      size_t base = (isMu ? (size_t)0 : (size_t)EM * EN) + (size_t)m * EN;
#pragma unroll
      for (int s = 0; s < 2; ++s)
#pragma unroll
        for (int v = 0; v < 8; ++v) {
          float val = (s ? c1[v] : c0[v]) + bb;
          int row = rowBase + r0 + s * 16 + v + rsel * 8;
          out[base + row] = isMu ? val : __expf(val);
        }
    }
  }
}

extern "C" void kernel_launch(void* const* d_in, const int* in_sizes, int n_in,
                              void* d_out, int out_size, void* d_ws, size_t ws_size,
                              hipStream_t stream) {
  const float* x    = (const float*)d_in[0];
  const float* W1   = (const float*)d_in[1];
  const float* b1   = (const float*)d_in[2];
  const float* Wh   = (const float*)d_in[3];
  const float* bh   = (const float*)d_in[4];
  const float* Wmu  = (const float*)d_in[5];
  const float* bmu  = (const float*)d_in[6];
  const float* Wsig = (const float*)d_in[7];
  const float* bsig = (const float*)d_in[8];
  float* out = (float*)d_out;

  dim3 grid(EN / ROWS, EM);  // (64, 100)
  size_t need = (size_t)TOTE * sizeof(_Float16);

  if (ws_size >= need) {
    char* ws = (char*)d_ws;
    _Float16* whH   = (_Float16*)ws;
    _Float16* w1H   = (_Float16*)(ws + (size_t)WHE * 2);
    _Float16* headH = (_Float16*)(ws + (size_t)(WHE + W1E) * 2);
    _Float16* xH    = (_Float16*)(ws + (size_t)(WHE + W1E + HDE) * 2);

    int pblocks = (TOTE + 255) / 256;
    prep_kernel<<<pblocks, 256, 0, stream>>>(x, W1, Wmu, Wsig, Wh,
                                             whH, w1H, headH, xH);
    ensemble_mlp_tdm_kernel<<<grid, 256, 0, stream>>>(b1, bh, bmu, bsig,
                                                      whH, w1H, headH, xH, out);
  } else {
    ensemble_mlp_kernel<<<grid, 256, 0, stream>>>(x, W1, b1, Wh, bh, Wmu, bmu,
                                                  Wsig, bsig, out);
  }
}
